// MultiHeadedAttention_40630390621070
// MI455X (gfx1250) — matrix operand order repaired and checked by reading, not compiled, not run on hardware
//
#include <hip/hip_runtime.h>

// ---------------------------------------------------------------------------
// MultiHeadedAttention for MI455X (gfx1250, wave32, WMMA 16x16x32 bf16).
// Pipeline:
//   1) fp32 -> bf16 conversion of X and the four weight matrices (one pass)
//   2) 3x bf16 GEMM (Q, K, V; V stored transposed per batch)
//   3) fused flash attention (bf16 WMMA, fp32 online softmax)
//   4) bf16 GEMM output projection -> fp32 d_out
// GEMMs double-buffer LDS and (when available) stage tiles with
// GLOBAL_LOAD_ASYNC_TO_LDS_B128 (ASYNCcnt), falling back to register staging.
// ---------------------------------------------------------------------------

typedef __attribute__((ext_vector_type(8)))  float   v8f;
typedef __attribute__((ext_vector_type(4)))  __bf16  v4bf;
typedef __attribute__((ext_vector_type(8)))  __bf16  v8bf;
typedef __attribute__((ext_vector_type(16))) __bf16  v16bf;

#define CAT16(lo, hi) \
  __builtin_shufflevector((lo), (hi), 0,1,2,3,4,5,6,7,8,9,10,11,12,13,14,15)
#define WMMA_BF16(a, b, c) \
  __builtin_amdgcn_wmma_f32_16x16x32_bf16(false, (a), false, (b), (short)0, (c), false, false)

#if defined(__has_builtin)
#if __has_builtin(__builtin_amdgcn_global_load_async_to_lds_b128)
#define USE_ASYNC_LDS 1
#endif
#endif
#ifndef USE_ASYNC_LDS
#define USE_ASYNC_LDS 0
#endif

#if USE_ASYNC_LDS
// Builtin signature (from compiler diagnostic): arg0 is
// `int __attribute__((vector_size(16))) addrspace(1)*`, arg1 the LDS-side
// pointer, then immediate offset and cache-policy.
typedef int b128i __attribute__((vector_size(16)));
typedef __attribute__((address_space(1))) b128i* gptr_b128;
typedef __attribute__((address_space(3))) b128i* lptr_b128;

__device__ __forceinline__ void async_copy_b128(const __bf16* g, __bf16* l) {
  __builtin_amdgcn_global_load_async_to_lds_b128(
      (gptr_b128)(void*)const_cast<__bf16*>(g),
      (lptr_b128)(void*)l, 0, 0);
}
__device__ __forceinline__ void wait_async0() {
#if __has_builtin(__builtin_amdgcn_s_wait_asynccnt)
  __builtin_amdgcn_s_wait_asynccnt(0);
#else
  asm volatile("s_wait_asynccnt 0x0" ::: "memory");
#endif
}
#endif

// ---------------------------------------------------------------------------
// fp32 -> bf16 elementwise conversion (n must be a multiple of 4).
// ---------------------------------------------------------------------------
__global__ __launch_bounds__(256) void cvt_f32_to_bf16(
    const float* __restrict__ in, __bf16* __restrict__ out, int n4)
{
  int i = blockIdx.x * 256 + threadIdx.x;
  if (i < n4) {
    float4 f = ((const float4*)in)[i];
    v4bf p; p.x = (__bf16)f.x; p.y = (__bf16)f.y; p.z = (__bf16)f.z; p.w = (__bf16)f.w;
    ((v4bf*)out)[i] = p;
  }
}

// ---------------------------------------------------------------------------
// GEMM: out[m,n] = sum_k A[m,k] * W[n,k] + bias[n]   (x @ W.T + b), bf16 in.
// Block tile 128(M) x 128(N) x 64(K); 256 threads = 8 waves (4 M-groups x
// 2 N-groups); each wave computes 32x64 via 8 accumulators. LDS double-buffered.
// ---------------------------------------------------------------------------
#define BM 128
#define BN 128
#define BK 64

__global__ __launch_bounds__(256) void gemm_bf16_wmma(
    const __bf16* __restrict__ Ab, const __bf16* __restrict__ Wb,
    const float* __restrict__ bias, void* __restrict__ out,
    int M, int N, int K, int out_is_bf16, int store_vt, int S)
{
  __shared__ __align__(16) __bf16 As[2][BM * BK]; // 2 x 16 KB
  __shared__ __align__(16) __bf16 Bs[2][BN * BK]; // 2 x 16 KB
  const int tid  = threadIdx.x;
  const int lane = tid & 31;
  const int wv   = tid >> 5;
  const int l16  = lane & 15;
  const bool hi  = lane >= 16;
  const int wm = wv & 3;   // 32-row group within block tile
  const int wn = wv >> 2;  // 64-col group within block tile
  const int m0 = blockIdx.x * BM;
  const int n0 = blockIdx.y * BN;

  // Staging map: 1024 b128 chunks (8 bf16 each) per tile, 4 per thread.
  int srow[4], scol[4];
  #pragma unroll
  for (int i = 0; i < 4; ++i) { int e = tid + i * 256; srow[i] = e >> 3; scol[i] = (e & 7) * 8; }

  v8f acc[2][4] = {};
  const int nk = K / BK;

#if !USE_ASYNC_LDS
  uint4 ra[4], rb[4];
#endif

  // ---- prologue: stage k-slab 0 into buffer 0 ----
#if USE_ASYNC_LDS
  #pragma unroll
  for (int i = 0; i < 4; ++i) {
    async_copy_b128(Ab + (size_t)(m0 + srow[i]) * K + scol[i], &As[0][srow[i] * BK + scol[i]]);
    async_copy_b128(Wb + (size_t)(n0 + srow[i]) * K + scol[i], &Bs[0][srow[i] * BK + scol[i]]);
  }
#else
  #pragma unroll
  for (int i = 0; i < 4; ++i) {
    ra[i] = *(const uint4*)(Ab + (size_t)(m0 + srow[i]) * K + scol[i]);
    rb[i] = *(const uint4*)(Wb + (size_t)(n0 + srow[i]) * K + scol[i]);
  }
  #pragma unroll
  for (int i = 0; i < 4; ++i) {
    *(uint4*)(&As[0][srow[i] * BK + scol[i]]) = ra[i];
    *(uint4*)(&Bs[0][srow[i] * BK + scol[i]]) = rb[i];
  }
#endif

  for (int ik = 0; ik < nk; ++ik) {
    const int cur = ik & 1, nxt = cur ^ 1;
#if USE_ASYNC_LDS
    wait_async0();
#endif
    __syncthreads();

    // Prefetch next k-slab into the other buffer while computing this one.
    const int k1 = (ik + 1) * BK;
    if (ik + 1 < nk) {
#if USE_ASYNC_LDS
      #pragma unroll
      for (int i = 0; i < 4; ++i) {
        async_copy_b128(Ab + (size_t)(m0 + srow[i]) * K + k1 + scol[i], &As[nxt][srow[i] * BK + scol[i]]);
        async_copy_b128(Wb + (size_t)(n0 + srow[i]) * K + k1 + scol[i], &Bs[nxt][srow[i] * BK + scol[i]]);
      }
#else
      #pragma unroll
      for (int i = 0; i < 4; ++i) {
        ra[i] = *(const uint4*)(Ab + (size_t)(m0 + srow[i]) * K + k1 + scol[i]);
        rb[i] = *(const uint4*)(Wb + (size_t)(n0 + srow[i]) * K + k1 + scol[i]);
      }
#endif
    }

    // Compute on current buffer: 2 x (6 fragment loads -> 8 WMMAs).
    #pragma unroll
    for (int kc = 0; kc < BK; kc += 32) {
      const int asel = hi ? 8 : 0;   // A: lanes<16 K{0..7,16..23}; >=16 K{8..15,24..31}
      const int bsel = hi ? 8 : 0;   // B: lanes<16 K{0..7,16..23}; >=16 K{8..15,24..31}
      v16bf af[2], bf4[4];
      #pragma unroll
      for (int mi = 0; mi < 2; ++mi) {
        const int arow = wm * 32 + mi * 16 + l16;
        v8bf lo = *(const v8bf*)(&As[cur][arow * BK + kc + asel]);
        v8bf h8 = *(const v8bf*)(&As[cur][arow * BK + kc + asel + 16]);
        af[mi] = CAT16(lo, h8);
      }
      #pragma unroll
      for (int t = 0; t < 4; ++t) {
        const int bn = wn * 64 + t * 16 + l16;
        v8bf lo = *(const v8bf*)(&Bs[cur][bn * BK + kc + bsel]);
        v8bf h8 = *(const v8bf*)(&Bs[cur][bn * BK + kc + bsel + 16]);
        bf4[t] = CAT16(lo, h8);
      }
      #pragma unroll
      for (int mi = 0; mi < 2; ++mi)
        #pragma unroll
        for (int t = 0; t < 4; ++t)
          acc[mi][t] = WMMA_BF16(af[mi], bf4[t], acc[mi][t]);
    }

#if !USE_ASYNC_LDS
    if (ik + 1 < nk) {
      #pragma unroll
      for (int i = 0; i < 4; ++i) {
        *(uint4*)(&As[nxt][srow[i] * BK + scol[i]]) = ra[i];
        *(uint4*)(&Bs[nxt][srow[i] * BK + scol[i]]) = rb[i];
      }
    }
#endif
  }

  // Epilogue: D layout -> lane<16: M=r; lane>=16: M=r+8; N = l16 within tile.
  const int rbase = m0 + wm * 32 + (hi ? 8 : 0);
  #pragma unroll
  for (int mi = 0; mi < 2; ++mi) {
    #pragma unroll
    for (int t = 0; t < 4; ++t) {
      const int col = n0 + wn * 64 + t * 16 + l16;
      const float bb = bias[col];
      #pragma unroll
      for (int r = 0; r < 8; ++r) {
        const int row = rbase + mi * 16 + r;
        float val = acc[mi][t][r] + bb;
        size_t idx;
        if (store_vt) {
          int bi = row / S, si = row - bi * S;              // m = b*S + s
          idx = (size_t)bi * N * S + (size_t)col * S + si;  // [b, n, s]
        } else {
          idx = (size_t)row * N + col;
        }
        if (out_is_bf16) ((__bf16*)out)[idx] = (__bf16)val;
        else             ((float*)out)[idx]  = val;
      }
    }
  }
}

// ---------------------------------------------------------------------------
// Fused flash attention. One wave per (b, h, 16-query tile). Streams keys in
// chunks of 32: K/V/mask loads issued up front (latency hidden by softmax),
// 4 WMMA for scores (K-dim = DH = 64), online softmax with shfl_xor row
// reductions over 16-lane halves, P relaid via LDS to A-fragment layout,
// 4 WMMA for PV (V pre-transposed so its columns are contiguous). bf16 out.
// ---------------------------------------------------------------------------
__global__ __launch_bounds__(128) void attn_fused_wmma(
    const __bf16* __restrict__ Qb, const __bf16* __restrict__ Kb,
    const __bf16* __restrict__ Vt, const int* __restrict__ mask,
    __bf16* __restrict__ atb, int Bn, int Sn, int Hn, int Dh)
{
  __shared__ __align__(16) __bf16 Pbuf[4][16 * 32];
  const int lane = threadIdx.x & 31;
  const int wv   = threadIdx.x >> 5;
  const int l16  = lane & 15;
  const bool hi  = lane >= 16;
  const int D    = Hn * Dh;

  const int qtiles = Sn >> 4;
  const int tile = blockIdx.x * 4 + wv;
  const int qt = tile % qtiles;
  const int h  = (tile / qtiles) % Hn;
  const int b  = tile / (qtiles * Hn);

  // Q fragments: two 16x32 A-fragments covering DH = 64.
  v16bf qa[2];
  {
    const int qrow = qt * 16 + l16;
    const __bf16* qp = Qb + ((size_t)(b * Sn + qrow)) * D + h * Dh;
    const int asel = hi ? 8 : 0;
    #pragma unroll
    for (int c = 0; c < 2; ++c) {
      v8bf lo = *(const v8bf*)(qp + c * 32 + asel);
      v8bf h8 = *(const v8bf*)(qp + c * 32 + asel + 16);
      qa[c] = CAT16(lo, h8);
    }
  }

  v8f acc[4] = {};
  float mrow[8], lrow[8];
  #pragma unroll
  for (int r = 0; r < 8; ++r) { mrow[r] = -1e30f; lrow[r] = 0.0f; }
  const float scale  = 0.125f;   // 1/sqrt(64)
  const float LOG2E  = 1.4426950408889634f;
  const float MASKED = -3.0e38f;

  __bf16* pb = Pbuf[wv];

  for (int k0 = 0; k0 < Sn; k0 += 32) {
    // ---- issue all global loads for this chunk up front ----
    const int bsel = hi ? 8 : 0;
    v16bf kf[2][2];
    #pragma unroll
    for (int kt = 0; kt < 2; ++kt) {
      const __bf16* kp = Kb + ((size_t)(b * Sn + k0 + kt * 16 + l16)) * D + h * Dh;
      #pragma unroll
      for (int cc = 0; cc < 2; ++cc) {
        v8bf lo = *(const v8bf*)(kp + cc * 32 + bsel);
        v8bf h8 = *(const v8bf*)(kp + cc * 32 + bsel + 16);
        kf[kt][cc] = CAT16(lo, h8);
      }
    }
    const int mk0 = mask[b * Sn + k0 + l16];
    const int mk1 = mask[b * Sn + k0 + 16 + l16];
    v16bf vf[4];
    const int vsel = hi ? 8 : 0;
    #pragma unroll
    for (int t = 0; t < 4; ++t) {
      const __bf16* vp = Vt + (size_t)b * D * Sn + (size_t)(h * Dh + t * 16 + l16) * Sn + k0 + vsel;
      v8bf lo = *(const v8bf*)(vp);
      v8bf h8 = *(const v8bf*)(vp + 16);
      vf[t] = CAT16(lo, h8);
    }

    // ---- scores 16x32 = Q(16x64) x K^T(64x32) ----
    v8f s[2];
    #pragma unroll
    for (int kt = 0; kt < 2; ++kt) {
      v8f c = {};
      c = WMMA_BF16(qa[0], kf[kt][0], c);
      c = WMMA_BF16(qa[1], kf[kt][1], c);
      s[kt] = c;
    }

    // ---- mask + scale + online softmax ----
    float rmax[8];
    #pragma unroll
    for (int r = 0; r < 8; ++r) {
      s[0][r] = mk0 ? s[0][r] * scale : MASKED;
      s[1][r] = mk1 ? s[1][r] * scale : MASKED;
      rmax[r] = fmaxf(s[0][r], s[1][r]);
    }
    #pragma unroll
    for (int off = 8; off > 0; off >>= 1)
      #pragma unroll
      for (int r = 0; r < 8; ++r)
        rmax[r] = fmaxf(rmax[r], __shfl_xor(rmax[r], off, 32));

    float alpha[8], psum[8];
    #pragma unroll
    for (int r = 0; r < 8; ++r) {
      const float mn = fmaxf(mrow[r], rmax[r]);
      alpha[r] = exp2f((mrow[r] - mn) * LOG2E);
      mrow[r] = mn;
      s[0][r] = exp2f((s[0][r] - mn) * LOG2E);
      s[1][r] = exp2f((s[1][r] - mn) * LOG2E);
      psum[r] = s[0][r] + s[1][r];
    }
    #pragma unroll
    for (int off = 8; off > 0; off >>= 1)
      #pragma unroll
      for (int r = 0; r < 8; ++r)
        psum[r] += __shfl_xor(psum[r], off, 32);
    #pragma unroll
    for (int r = 0; r < 8; ++r)
      lrow[r] = lrow[r] * alpha[r] + psum[r];
    #pragma unroll
    for (int t = 0; t < 4; ++t)
      #pragma unroll
      for (int r = 0; r < 8; ++r)
        acc[t][r] *= alpha[r];

    // ---- P (C/D layout) -> LDS -> reload in A-fragment layout ----
    const int prow = hi ? 8 : 0;
    #pragma unroll
    for (int r = 0; r < 8; ++r) {
      pb[(prow + r) * 32 + l16]      = (__bf16)s[0][r];
      pb[(prow + r) * 32 + 16 + l16] = (__bf16)s[1][r];
    }
    const int asel = hi ? 8 : 0;
    v8bf plo = *(const v8bf*)(pb + l16 * 32 + asel);
    v8bf phi = *(const v8bf*)(pb + l16 * 32 + asel + 16);
    v16bf pf = CAT16(plo, phi);

    // ---- PV: acc(16x64) += P(16x32) x V(32x64) ----
    #pragma unroll
    for (int t = 0; t < 4; ++t)
      acc[t] = WMMA_BF16(pf, vf[t], acc[t]);
  }

  // ---- normalize and store bf16 attended in merged (B,S,D) layout ----
  const int rowbase = qt * 16 + (hi ? 8 : 0);
  #pragma unroll
  for (int t = 0; t < 4; ++t) {
    const int col = h * Dh + t * 16 + l16;
    #pragma unroll
    for (int r = 0; r < 8; ++r) {
      const int row = rowbase + r;
      atb[((size_t)(b * Sn + row)) * D + col] = (__bf16)(acc[t][r] / lrow[r]);
    }
  }
}

// ---------------------------------------------------------------------------
extern "C" void kernel_launch(void* const* d_in, const int* in_sizes, int n_in,
                              void* d_out, int out_size, void* d_ws, size_t ws_size,
                              hipStream_t stream) {
  (void)in_sizes; (void)n_in; (void)out_size; (void)ws_size;
  const float* X  = (const float*)d_in[0];
  const int*   mk = (const int*)  d_in[1];
  const float* Wq = (const float*)d_in[2];
  const float* bq = (const float*)d_in[3];
  const float* Wk = (const float*)d_in[4];
  const float* bk = (const float*)d_in[5];
  const float* Wv = (const float*)d_in[6];
  const float* bv = (const float*)d_in[7];
  const float* Wo = (const float*)d_in[8];
  const float* bo = (const float*)d_in[9];

  const int B = 2, S = 2048, H = 16, DH = 64;
  const int D = H * DH, M = B * S;
  const size_t MD = (size_t)M * D;   // 4M elements
  const size_t DD = (size_t)D * D;   // 1M elements

  // Workspace (bf16): Xb | Wqb | Wkb | Wvb | Wob | Qb | Kb | Vt | atb  (~48 MB)
  __bf16* Xb  = (__bf16*)d_ws;
  __bf16* Wqb = Xb  + MD;
  __bf16* Wkb = Wqb + DD;
  __bf16* Wvb = Wkb + DD;
  __bf16* Wob = Wvb + DD;
  __bf16* Qb  = Wob + DD;
  __bf16* Kb  = Qb  + MD;
  __bf16* Vt  = Kb  + MD;
  __bf16* atb = Vt  + MD;

  // 1) fp32 -> bf16 conversions
  cvt_f32_to_bf16<<<(int)(MD / 4 / 256), 256, 0, stream>>>(X,  Xb,  (int)(MD / 4));
  cvt_f32_to_bf16<<<(int)(DD / 4 / 256), 256, 0, stream>>>(Wq, Wqb, (int)(DD / 4));
  cvt_f32_to_bf16<<<(int)(DD / 4 / 256), 256, 0, stream>>>(Wk, Wkb, (int)(DD / 4));
  cvt_f32_to_bf16<<<(int)(DD / 4 / 256), 256, 0, stream>>>(Wv, Wvb, (int)(DD / 4));
  cvt_f32_to_bf16<<<(int)(DD / 4 / 256), 256, 0, stream>>>(Wo, Wob, (int)(DD / 4));

  // 2) QKV projections (V transposed per batch for contiguous PV columns)
  dim3 gg(M / BM, D / BN);
  gemm_bf16_wmma<<<gg, 256, 0, stream>>>(Xb, Wqb, bq, (void*)Qb, M, D, D, 1, 0, S);
  gemm_bf16_wmma<<<gg, 256, 0, stream>>>(Xb, Wkb, bk, (void*)Kb, M, D, D, 1, 0, S);
  gemm_bf16_wmma<<<gg, 256, 0, stream>>>(Xb, Wvb, bv, (void*)Vt, M, D, D, 1, 1, S);

  // 3) fused attention
  attn_fused_wmma<<<(B * H * (S / 16)) / 4, 128, 0, stream>>>(Qb, Kb, Vt, mk, atb, B, S, H, DH);

  // 4) output projection -> fp32 d_out
  gemm_bf16_wmma<<<gg, 256, 0, stream>>>(atb, Wob, bo, d_out, M, D, D, 0, 0, S);
}
